// CustomMultiheadAttention_27195732918512
// MI455X (gfx1250) — compile-verified, matrix operand order checked by reading, not run on hardware
//
#include <hip/hip_runtime.h>

// ---------------------------------------------------------------------------
// CDNA5 (gfx1250) multi-head attention forward.
// All matmuls on v_wmma_f32_16x16x32_bf16 (f32 accumulate); f32 -> bf16 RNE
// conversion on the fly (native v_cvt_pk_bf16_f32 when available).
// wave32 fragment layouts per ISA 7.12.2.
// ---------------------------------------------------------------------------

#define DEV static __device__ __forceinline__

typedef __attribute__((ext_vector_type(16))) __bf16 bf16x16;
typedef __attribute__((ext_vector_type(2)))  __bf16 bf16x2;
typedef __attribute__((ext_vector_type(8)))  float  f32x8;

struct alignas(16) u32x4 { unsigned x, y, z, w; };
struct U32x8 { u32x4 a, b; };   // 32 bytes == one bf16x16 fragment

DEV bf16x16 as_bf16x16(U32x8 x) { return __builtin_bit_cast(bf16x16, x); }

// f32 -> bf16 round-to-nearest-even (bit-trick fallback)
DEV unsigned f2bf_raw(float f) {
  unsigned u = __builtin_bit_cast(unsigned, f);
  u += 0x7FFFu + ((u >> 16) & 1u);
  return u >> 16;
}

// Pack two f32 into two bf16 (one v_cvt_pk_bf16_f32 when the builtin exists).
DEV unsigned pack2(float lo, float hi) {
#if __has_builtin(__builtin_amdgcn_cvt_pk_bf16_f32)
  bf16x2 p = __builtin_amdgcn_cvt_pk_bf16_f32(lo, hi);
  return __builtin_bit_cast(unsigned, p);
#else
  return f2bf_raw(lo) | (f2bf_raw(hi) << 16);
#endif
}

DEV unsigned short f2bf16(float f) {
#if __has_builtin(__builtin_amdgcn_cvt_pk_bf16_f32)
  return (unsigned short)(pack2(f, 0.0f) & 0xFFFFu);
#else
  return (unsigned short)f2bf_raw(f);
#endif
}

DEV f32x8 wmma_bf16(bf16x16 a, bf16x16 b, f32x8 c) {
  // (neg_a, A, neg_b, B, c_mod, C, reuse_a, reuse_b)
  return __builtin_amdgcn_wmma_f32_16x16x32_bf16(false, a, false, b, (short)0, c,
                                                 false, false);
}

constexpr int cS = 2048, cB = 2, cE = 1024, cH = 16, cD = 64;
constexpr int BM = 128, BN = 128, BK = 32;

// A-fragment (16x32 bf16) loader from an LDS tile laid out [row][BK].
// Lane l: row = base + (l&15); K-chunks at (l>>4)*8 and +16 (ISA 7.12.2).
DEV bf16x16 frag_lds(const unsigned short* tile, int row16, int lane) {
  const int r  = row16 + (lane & 15);
  const int kh = (lane >> 4) * 8;
  const unsigned short* p = tile + r * BK;
  U32x8 f;
  f.a = *(const u32x4*)(p + kh);
  f.b = *(const u32x4*)(p + kh + 16);
  return as_bf16x16(f);
}

// Fragment from a global bf16 row (contraction contiguous), rowp = lane's row.
DEV bf16x16 frag_g(const unsigned short* rowp, int kh) {
  U32x8 f;
  f.a = *(const u32x4*)(rowp + kh);
  f.b = *(const u32x4*)(rowp + kh + 16);
  return as_bf16x16(f);
}

// ---------------------------------------------------------------------------
// Kernel 1: Q/K/V projections.  Y[r,n] = X[r,:]·W[n,:] + b[n],
// r = s*B+b (4096 rows), n = h*64+d.  Writes bf16:
//   sel 0/1 -> Q/K as [b,h,s,d];  sel 2 -> V transposed as [b,h,d,s].
// ---------------------------------------------------------------------------
__global__ __launch_bounds__(256) void k_proj_qkv(
    const float* __restrict__ Xq, const float* __restrict__ Xk,
    const float* __restrict__ Xv,
    const float* __restrict__ Wq, const float* __restrict__ bq,
    const float* __restrict__ Wk, const float* __restrict__ bk,
    const float* __restrict__ Wv, const float* __restrict__ bv,
    unsigned short* __restrict__ Qb, unsigned short* __restrict__ Kb,
    unsigned short* __restrict__ Vt) {
  __shared__ alignas(16) unsigned short lA[BM * BK];
  __shared__ alignas(16) unsigned short lB[BN * BK];

  const int sel = blockIdx.z;
  const float* X    = (sel == 0) ? Xq : (sel == 1) ? Xk : Xv;
  const float* W    = (sel == 0) ? Wq : (sel == 1) ? Wk : Wv;
  const float* bias = (sel == 0) ? bq : (sel == 1) ? bk : bv;

  const int m0 = blockIdx.y * BM;
  const int n0 = blockIdx.x * BN;
  const int tid = threadIdx.x;
  const int lane = tid & 31, wid = tid >> 5;
  const int wm = (wid & 3) * 32;   // 4 waves along M
  const int wn = (wid >> 2) * 64;  // 2 waves along N

  f32x8 acc[2][4] = {};

  const int ldr = tid >> 1;        // 0..127 tile row
  const int ldc = (tid & 1) * 16;  // 0 / 16

  for (int k0 = 0; k0 < cE; k0 += BK) {
    const float* pa = X + (size_t)(m0 + ldr) * cE + k0 + ldc;
    const float* pb = W + (size_t)(n0 + ldr) * cE + k0 + ldc;
    if (k0 + BK < cE) {  // prefetch next K tile into WGP$ -> global_prefetch_b8
      __builtin_prefetch(pa + BK, 0, 3);
      __builtin_prefetch(pb + BK, 0, 3);
    }
    float4 a0 = *(const float4*)pa,       a1 = *(const float4*)(pa + 4);
    float4 a2 = *(const float4*)(pa + 8), a3 = *(const float4*)(pa + 12);
    float4 b0 = *(const float4*)pb,       b1 = *(const float4*)(pb + 4);
    float4 b2 = *(const float4*)(pb + 8), b3 = *(const float4*)(pb + 12);
    __syncthreads();
    u32x4* da = (u32x4*)&lA[ldr * BK + ldc];
    da[0] = u32x4{pack2(a0.x, a0.y), pack2(a0.z, a0.w), pack2(a1.x, a1.y), pack2(a1.z, a1.w)};
    da[1] = u32x4{pack2(a2.x, a2.y), pack2(a2.z, a2.w), pack2(a3.x, a3.y), pack2(a3.z, a3.w)};
    u32x4* db = (u32x4*)&lB[ldr * BK + ldc];
    db[0] = u32x4{pack2(b0.x, b0.y), pack2(b0.z, b0.w), pack2(b1.x, b1.y), pack2(b1.z, b1.w)};
    db[1] = u32x4{pack2(b2.x, b2.y), pack2(b2.z, b2.w), pack2(b3.x, b3.y), pack2(b3.z, b3.w)};
    __syncthreads();

    bf16x16 af[2], bfr[4];
#pragma unroll
    for (int i = 0; i < 2; ++i) af[i] = frag_lds(lA, wm + i * 16, lane);
#pragma unroll
    for (int j = 0; j < 4; ++j) bfr[j] = frag_lds(lB, wn + j * 16, lane);
#pragma unroll
    for (int i = 0; i < 2; ++i)
#pragma unroll
      for (int j = 0; j < 4; ++j) acc[i][j] = wmma_bf16(af[i], bfr[j], acc[i][j]);
  }

  unsigned short* outp = (sel == 0) ? Qb : (sel == 1) ? Kb : Vt;
  const int colOff = lane & 15;
  const int rowHalf = (lane >> 4) * 8;
#pragma unroll
  for (int i = 0; i < 2; ++i) {
#pragma unroll
    for (int j = 0; j < 4; ++j) {
      const int col = n0 + wn + j * 16 + colOff;
      const float bcol = bias[col];
      const int h = col >> 6, d = col & 63;
#pragma unroll
      for (int v = 0; v < 8; ++v) {
        const int r = m0 + wm + i * 16 + rowHalf + v;  // r = s*B + b
        const int s = r >> 1, b = r & 1;
        const float val = acc[i][j][v] + bcol;
        size_t idx;
        if (sel == 2) idx = (((size_t)b * cH + h) * cD + d) * cS + s;  // V^T
        else          idx = (((size_t)b * cH + h) * cS + s) * cD + d;  // Q,K
        outp[idx] = f2bf16(val);
      }
    }
  }
}

// ---------------------------------------------------------------------------
// Kernel 2: scores = (Q·K^T)*0.125 + bias, row softmax, write attn f32.
// One WG per (b*h, 16-row q block).  Each of 8 waves owns 16 column tiles
// (full 2048-wide row resident in 128 accumulator VGPRs per lane).
// ---------------------------------------------------------------------------
__global__ __launch_bounds__(256) void k_attn_softmax(
    const unsigned short* __restrict__ Qb, const unsigned short* __restrict__ Kb,
    const float* __restrict__ bias_matrix, float* __restrict__ attn) {
  const int qblk = blockIdx.x;  // 0..127
  const int bh   = blockIdx.y;  // 0..31
  const int tid = threadIdx.x, lane = tid & 31, wid = tid >> 5;
  const int q0 = qblk * 16;
  __shared__ float wred[8][16];

  const int m  = lane & 15;
  const int kh = (lane >> 4) * 8;
  const int colOff  = lane & 15;
  const int rowHalf = (lane >> 4) * 8;

  const unsigned short* Qrow  = Qb + ((size_t)bh * cS + q0 + m) * cD;
  const unsigned short* Kbase = Kb + (size_t)bh * cS * cD;

  bf16x16 a[2];
#pragma unroll
  for (int ks = 0; ks < 2; ++ks) a[ks] = frag_g(Qrow + ks * 32, kh);

  f32x8 acc[16] = {};
#pragma unroll
  for (int t = 0; t < 16; ++t) {
    const int j = wid + t * 8;  // column tile 0..127
    const unsigned short* Krow = Kbase + (size_t)(j * 16 + m) * cD;
#pragma unroll
    for (int ks = 0; ks < 2; ++ks)
      acc[t] = wmma_bf16(a[ks], frag_g(Krow + ks * 32, kh), acc[t]);
  }

  const float scale = 0.125f;  // D^-0.5
#pragma unroll
  for (int t = 0; t < 16; ++t) {
    const int col = (wid + t * 8) * 16 + colOff;
#pragma unroll
    for (int v = 0; v < 8; ++v)
      acc[t][v] = acc[t][v] * scale +
                  bias_matrix[(size_t)(q0 + rowHalf + v) * cS + col];
  }

  // ---- row max: per-lane over tiles, butterfly over the 16-lane group ----
  float rmax[8];
#pragma unroll
  for (int v = 0; v < 8; ++v) {
    float mv = acc[0][v];
#pragma unroll
    for (int t = 1; t < 16; ++t) mv = fmaxf(mv, acc[t][v]);
    for (int o = 8; o >= 1; o >>= 1) mv = fmaxf(mv, __shfl_xor(mv, o));
    if (colOff == 0) wred[wid][rowHalf + v] = mv;
    rmax[v] = mv;
  }
  __syncthreads();
#pragma unroll
  for (int v = 0; v < 8; ++v) {
    float x = wred[0][rowHalf + v];
#pragma unroll
    for (int w = 1; w < 8; ++w) x = fmaxf(x, wred[w][rowHalf + v]);
    rmax[v] = x;
  }
  __syncthreads();

  // ---- exp + row sum ----
  float psum[8];
#pragma unroll
  for (int v = 0; v < 8; ++v) psum[v] = 0.f;
#pragma unroll
  for (int t = 0; t < 16; ++t)
#pragma unroll
    for (int v = 0; v < 8; ++v) {
      const float e = __expf(acc[t][v] - rmax[v]);
      acc[t][v] = e;
      psum[v] += e;
    }
#pragma unroll
  for (int v = 0; v < 8; ++v) {
    for (int o = 8; o >= 1; o >>= 1) psum[v] += __shfl_xor(psum[v], o);
    if (colOff == 0) wred[wid][rowHalf + v] = psum[v];
  }
  __syncthreads();
  float rinv[8];
#pragma unroll
  for (int v = 0; v < 8; ++v) {
    float s = 0.f;
#pragma unroll
    for (int w = 0; w < 8; ++w) s += wred[w][rowHalf + v];
    rinv[v] = 1.0f / s;
  }

  // ---- normalize + store attn [b,h,q,k] f32 ----
  float* outr = attn + ((size_t)bh * cS + q0) * cS;
#pragma unroll
  for (int t = 0; t < 16; ++t) {
    const int col = (wid + t * 8) * 16 + colOff;
#pragma unroll
    for (int v = 0; v < 8; ++v)
      outr[(size_t)(rowHalf + v) * cS + col] = acc[t][v] * rinv[v];
  }
}

// ---------------------------------------------------------------------------
// Kernel 3: context = attn · V   (per b,h: [2048x2048]·[2048x64]).
// attn read f32 -> bf16 in-register; V pre-transposed [b,h,d,s] so B-fragments
// load straight from global.  Each wave: one 16-row q tile x all 4 d tiles.
// ---------------------------------------------------------------------------
__global__ __launch_bounds__(256) void k_context(
    const float* __restrict__ attn, const unsigned short* __restrict__ Vt,
    unsigned short* __restrict__ ctx) {
  const int bh = blockIdx.y;
  const int b = bh >> 4, h = bh & 15;
  const int tid = threadIdx.x, lane = tid & 31, wid = tid >> 5;
  const int q0 = blockIdx.x * 128 + wid * 16;
  const int m = lane & 15, kh = (lane >> 4) * 8;

  const float* Arow = attn + ((size_t)bh * cS + q0 + m) * cS;
  const unsigned short* Bbase = Vt + (size_t)bh * cD * cS;  // [d][s]

  f32x8 acc[4] = {};
  for (int k0 = 0; k0 < cS; k0 += 32) {
    const float* pa = Arow + k0 + kh;
    if (k0 + 32 < cS) __builtin_prefetch(pa + 32, 0, 3);
    float4 c00 = *(const float4*)pa,        c01 = *(const float4*)(pa + 4);
    float4 c10 = *(const float4*)(pa + 16), c11 = *(const float4*)(pa + 20);
    U32x8 f;
    f.a = u32x4{pack2(c00.x, c00.y), pack2(c00.z, c00.w),
                pack2(c01.x, c01.y), pack2(c01.z, c01.w)};
    f.b = u32x4{pack2(c10.x, c10.y), pack2(c10.z, c10.w),
                pack2(c11.x, c11.y), pack2(c11.z, c11.w)};
    const bf16x16 a = as_bf16x16(f);
#pragma unroll
    for (int j = 0; j < 4; ++j) {
      const unsigned short* pb = Bbase + (size_t)(j * 16 + m) * cS + k0;
      acc[j] = wmma_bf16(a, frag_g(pb, kh), acc[j]);
    }
  }

  const int colOff = lane & 15, rowHalf = (lane >> 4) * 8;
#pragma unroll
  for (int j = 0; j < 4; ++j) {
    const int d = j * 16 + colOff;
#pragma unroll
    for (int v = 0; v < 8; ++v) {
      const int qq = q0 + rowHalf + v;
      ctx[((size_t)b * cS + qq) * cE + h * cD + d] = f2bf16(acc[j][v]);
    }
  }
}

// ---------------------------------------------------------------------------
// Kernel 4: out = ctx · Wo^T + bo, permuted store to [S,B,E] f32.
// ---------------------------------------------------------------------------
__global__ __launch_bounds__(256) void k_out_proj(
    const unsigned short* __restrict__ Xb, const float* __restrict__ Wo,
    const float* __restrict__ bo, float* __restrict__ out) {
  __shared__ alignas(16) unsigned short lA[BM * BK];
  __shared__ alignas(16) unsigned short lB[BN * BK];

  const int m0 = blockIdx.y * BM;
  const int n0 = blockIdx.x * BN;
  const int tid = threadIdx.x;
  const int lane = tid & 31, wid = tid >> 5;
  const int wm = (wid & 3) * 32;
  const int wn = (wid >> 2) * 64;

  f32x8 acc[2][4] = {};
  const int ldr = tid >> 1;
  const int ldc = (tid & 1) * 16;

  for (int k0 = 0; k0 < cE; k0 += BK) {
    const unsigned short* pa = Xb + (size_t)(m0 + ldr) * cE + k0 + ldc;
    const float* pb = Wo + (size_t)(n0 + ldr) * cE + k0 + ldc;
    if (k0 + BK < cE) {
      __builtin_prefetch(pa + BK, 0, 3);
      __builtin_prefetch(pb + BK, 0, 3);
    }
    u32x4 a0 = *(const u32x4*)pa, a1 = *(const u32x4*)(pa + 8);
    float4 b0 = *(const float4*)pb,       b1 = *(const float4*)(pb + 4);
    float4 b2 = *(const float4*)(pb + 8), b3 = *(const float4*)(pb + 12);
    __syncthreads();
    u32x4* da = (u32x4*)&lA[ldr * BK + ldc];
    da[0] = a0;
    da[1] = a1;
    u32x4* db = (u32x4*)&lB[ldr * BK + ldc];
    db[0] = u32x4{pack2(b0.x, b0.y), pack2(b0.z, b0.w), pack2(b1.x, b1.y), pack2(b1.z, b1.w)};
    db[1] = u32x4{pack2(b2.x, b2.y), pack2(b2.z, b2.w), pack2(b3.x, b3.y), pack2(b3.z, b3.w)};
    __syncthreads();

    bf16x16 af[2], bfr[4];
#pragma unroll
    for (int i = 0; i < 2; ++i) af[i] = frag_lds(lA, wm + i * 16, lane);
#pragma unroll
    for (int j = 0; j < 4; ++j) bfr[j] = frag_lds(lB, wn + j * 16, lane);
#pragma unroll
    for (int i = 0; i < 2; ++i)
#pragma unroll
      for (int j = 0; j < 4; ++j) acc[i][j] = wmma_bf16(af[i], bfr[j], acc[i][j]);
  }

  const int colOff = lane & 15;
  const int rowHalf = (lane >> 4) * 8;
#pragma unroll
  for (int i = 0; i < 2; ++i) {
#pragma unroll
    for (int j = 0; j < 4; ++j) {
      const int col = n0 + wn + j * 16 + colOff;
      const float bcol = bo[col];
#pragma unroll
      for (int v = 0; v < 8; ++v) {
        const int r = m0 + wm + i * 16 + rowHalf + v;  // r = b*S + q
        const int b = r >> 11, q = r & (cS - 1);
        out[((size_t)q * cB + b) * cE + col] = acc[i][j][v] + bcol;
      }
    }
  }
}

// ---------------------------------------------------------------------------
extern "C" void kernel_launch(void* const* d_in, const int* in_sizes, int n_in,
                              void* d_out, int out_size, void* d_ws,
                              size_t ws_size, hipStream_t stream) {
  const float* query = (const float*)d_in[0];
  const float* key_  = (const float*)d_in[1];
  const float* value = (const float*)d_in[2];
  const float* Wq = (const float*)d_in[3];
  const float* bq = (const float*)d_in[4];
  const float* Wk = (const float*)d_in[5];
  const float* bk = (const float*)d_in[6];
  const float* Wv = (const float*)d_in[7];
  const float* bv = (const float*)d_in[8];
  const float* Wo = (const float*)d_in[9];
  const float* bo = (const float*)d_in[10];
  const float* bias_matrix = (const float*)d_in[11];

  float* out  = (float*)d_out;
  float* attn = out + (size_t)cS * cB * cE;  // outputs: [out | attn]

  const size_t nQKV = (size_t)cB * cH * cS * cD;  // 4M bf16 elems each
  unsigned short* Qb  = (unsigned short*)d_ws;
  unsigned short* Kb  = Qb + nQKV;
  unsigned short* Vt  = Kb + nQKV;                // V transposed [b,h,d,s]
  unsigned short* ctx = Vt + nQKV;                // [b,s,e] bf16

  const dim3 blk(256);
  const dim3 g1(cE / BN, (cS * cB) / BM, 3);
  k_proj_qkv<<<g1, blk, 0, stream>>>(query, key_, value, Wq, bq, Wk, bk, Wv, bv,
                                     Qb, Kb, Vt);
  const dim3 g2(cS / 16, cB * cH);
  k_attn_softmax<<<g2, blk, 0, stream>>>(Qb, Kb, bias_matrix, attn);
  const dim3 g3(cS / 128, cB * cH);
  k_context<<<g3, blk, 0, stream>>>(attn, Vt, ctx);
  const dim3 g4(cE / BN, (cB * cS) / BM);
  k_out_proj<<<g4, blk, 0, stream>>>(ctx, Wo, bo, out);
}
